// StateRN_77154792505505
// MI455X (gfx1250) — matrix operand order repaired and checked by reading, not compiled, not run on hardware
//
#include <hip/hip_runtime.h>

// ---------------------------------------------------------------------------
// CDNA5 (gfx1250) fused Relation Network forward
//   g-MLP (4x Linear+ReLU, bf16 WMMA) fused with pair construction + sum-pool
//   f-MLP via same WMMA tiling; f3 + log_softmax per-row wave kernel
// ---------------------------------------------------------------------------

typedef __attribute__((ext_vector_type(16))) __bf16       v16bf;
typedef __attribute__((ext_vector_type(8)))  float        v8f;
typedef __attribute__((ext_vector_type(4)))  unsigned int v4u;

union Frag { v16bf v; v4u q[2]; };

static __device__ __forceinline__ v8f wmma_bf16(const Frag& a, const Frag& b, v8f c) {
  return __builtin_amdgcn_wmma_f32_16x16x32_bf16(false, a.v, false, b.v,
                                                 (short)0, c, false, false);
}

// ---------------------------------------------------------------------------
// Weight swizzle: f32 [K][N] row-major  ->  bf16 WMMA B-fragment tiles.
// Tile (kt,nt) covers K[kt*32..+32) x N[nt*16..+16); stored as 32 lanes x
// 16 halves (32B) contiguous, in the contiguous k order (k = 16h + j, not AMD's
// table order) of the B operand, matching the A loads:
//   lane<16 : N = nt*16+lane,      K = kt*32 + j      (j = 2v+e = 0..15)
//   lane>=16: N = nt*16+(lane-16), K = kt*32 + 16 + j
// ---------------------------------------------------------------------------
__global__ __launch_bounds__(256)
void swizzle_w(const float* __restrict__ W, __bf16* __restrict__ dst,
               int Kpad, int N, int Kact)
{
  const int NT = N >> 4;
  const int total = (Kpad >> 5) * NT * 512;
  int idx = blockIdx.x * 256 + threadIdx.x;
  if (idx >= total) return;
  int tile = idx >> 9;
  int wi   = idx & 511;
  int lane = wi >> 4;
  int j    = wi & 15;
  int kt = tile / NT, nt = tile - kt * NT;
  int nloc = lane & 15, kgrp = lane >> 4;
  int K = kt * 32 + kgrp * 16 + j;
  int Ncol = nt * 16 + nloc;
  float v = (K < Kact) ? W[(size_t)K * N + Ncol] : 0.f;
  dst[idx] = (__bf16)v;
}

__global__ __launch_bounds__(256)
void zero_f32(float* __restrict__ p, int n)
{
  int i = blockIdx.x * 256 + threadIdx.x;
  if (i < n) p[i] = 0.f;
}

// ---------------------------------------------------------------------------
// One fused g-layer: OUT[128][512] = relu(IN[128][ldk] @ W + bias), bf16 LDS.
// 8 waves; wave w owns col-tiles [w*4, w*4+4), all 8 row-tiles (acc[8]).
// Each weight tile is loaded exactly once per workgroup.
// ---------------------------------------------------------------------------
static __device__ __forceinline__
void gemm_layer(const __bf16* __restrict__ in, int ldk, int ktiles,
                const __bf16* __restrict__ wsw, const float* __restrict__ bias,
                __bf16* __restrict__ out)
{
  const int tid  = threadIdx.x;
  const int lane = tid & 31, wave = tid >> 5;
  const int sel  = lane >> 4, nl = lane & 15;
  const v4u* wbase = (const v4u*)wsw;

  for (int ci = 0; ci < 4; ++ci) {
    const int nt = wave * 4 + ci;
    v8f acc[8] = {};
    for (int kt = 0; kt < ktiles; ++kt) {
      Frag bw;
      const v4u* wp = wbase + ((size_t)(kt * 32 + nt) * 32 + lane) * 2;
      bw.q[0] = wp[0]; bw.q[1] = wp[1];
      #pragma unroll
      for (int m = 0; m < 8; ++m) {
        Frag a;
        const v4u* ap = (const v4u*)(in + (m * 16 + nl) * ldk + kt * 32 + sel * 16);
        a.q[0] = ap[0]; a.q[1] = ap[1];   // K-chunks at +0B and +16B
        acc[m] = wmma_bf16(a, bw, acc[m]);
      }
    }
    const int col = nt * 16 + nl;
    const float bv = bias[col];
    #pragma unroll
    for (int m = 0; m < 8; ++m) {
      #pragma unroll
      for (int r = 0; r < 8; ++r) {
        float v = acc[m][r] + bv;
        v = v > 0.f ? v : 0.f;
        out[(m * 16 + sel * 8 + r) * 512 + col] = (__bf16)v;
      }
    }
  }
}

// ---------------------------------------------------------------------------
// Fused g-MLP + sum-pool. grid = (5, 1024): blockIdx.y = batch, blockIdx.x =
// 128-pair tile (625 pairs padded to 640; padded rows discarded at pooling).
// Dynamic LDS: pair[128*32] + buf0[128*512] + buf1[128*512] bf16 = 264 KB.
// ---------------------------------------------------------------------------
__global__ __launch_bounds__(256)
void g_mlp_pool(const float* __restrict__ img, const float* __restrict__ qst,
                const __bf16* __restrict__ swg1, const float* __restrict__ g1b,
                const __bf16* __restrict__ swg2, const float* __restrict__ g2b,
                const __bf16* __restrict__ swg3, const float* __restrict__ g3b,
                const __bf16* __restrict__ swg4, const float* __restrict__ g4b,
                float* __restrict__ xg)
{
  extern __shared__ char smem_raw[];
  __bf16* pairb = (__bf16*)smem_raw;       // 128*32
  __bf16* buf0  = pairb + 128 * 32;        // 128*512
  __bf16* buf1  = buf0 + 128 * 512;        // 128*512
  const int t = blockIdx.x, b = blockIdx.y;
  const int tid = threadIdx.x;

  // Build pair features: row pr = t*128+row -> (p=pr/25, q=pr%25)
  // feat[0:6]=x_flat[b,q,:], feat[6:12]=x_flat[b,p,:], feat[12:23]=qst[b]
  #pragma unroll
  for (int e = 0; e < 16; ++e) {
    int idx = e * 256 + tid;
    int row = idx >> 5, k = idx & 31;
    int pr = t * 128 + row;
    int p = pr / 25, q = pr - p * 25;
    float v = 0.f;
    if (p < 25) {
      if (k < 6)       v = img[(b * 6 + k) * 25 + q];
      else if (k < 12) v = img[(b * 6 + (k - 6)) * 25 + p];
      else if (k < 23) v = qst[b * 11 + (k - 12)];
    }
    pairb[idx] = (__bf16)v;
  }
  __syncthreads();

  gemm_layer(pairb, 32, 1,  swg1, g1b, buf0); __syncthreads();
  gemm_layer(buf0, 512, 16, swg2, g2b, buf1); __syncthreads();
  gemm_layer(buf1, 512, 16, swg3, g3b, buf0); __syncthreads();
  gemm_layer(buf0, 512, 16, swg4, g4b, buf1); __syncthreads();

  // Sum-pool valid rows of this tile into xg[b][:]
  int valid = 625 - t * 128; if (valid > 128) valid = 128;
  for (int c = tid; c < 512; c += 256) {
    float s = 0.f;
    for (int r = 0; r < valid; ++r) s += (float)buf1[r * 512 + c];
    atomicAdd(&xg[(size_t)b * 512 + c], s);
  }
}

// ---------------------------------------------------------------------------
// f-layer GEMM: C[128rows][N] = (relu)(A @ W + bias), A f32 global staged to
// bf16 LDS, same WMMA tiling. grid.x = Mtotal/128.
// ---------------------------------------------------------------------------
__global__ __launch_bounds__(256)
void gemm_f(const float* __restrict__ A, const __bf16* __restrict__ wsw,
            const float* __restrict__ bias, float* __restrict__ C,
            int K, int N, int relu)
{
  extern __shared__ char smem_raw[];
  __bf16* ain = (__bf16*)smem_raw;          // 128*K
  const int m_base = blockIdx.x * 128;
  const int tid = threadIdx.x;
  const int nk = 128 * K;
  for (int i = tid; i < nk; i += 256)
    ain[i] = (__bf16)A[(size_t)m_base * K + i];
  __syncthreads();

  const int lane = tid & 31, wave = tid >> 5;
  const int sel = lane >> 4, nl = lane & 15;
  const int NT = N >> 4, ktiles = K >> 5;
  const int cpw = NT >> 3;
  const v4u* wbase = (const v4u*)wsw;

  for (int ci = 0; ci < cpw; ++ci) {
    const int nt = wave * cpw + ci;
    v8f acc[8] = {};
    for (int kt = 0; kt < ktiles; ++kt) {
      Frag bw;
      const v4u* wp = wbase + ((size_t)(kt * NT + nt) * 32 + lane) * 2;
      bw.q[0] = wp[0]; bw.q[1] = wp[1];
      #pragma unroll
      for (int m = 0; m < 8; ++m) {
        Frag a;
        const v4u* ap = (const v4u*)(ain + (m * 16 + nl) * K + kt * 32 + sel * 16);
        a.q[0] = ap[0]; a.q[1] = ap[1];
        acc[m] = wmma_bf16(a, bw, acc[m]);
      }
    }
    const int col = nt * 16 + nl;
    const float bv = bias[col];
    #pragma unroll
    for (int m = 0; m < 8; ++m) {
      #pragma unroll
      for (int r = 0; r < 8; ++r) {
        float v = acc[m][r] + bv;
        if (relu) v = v > 0.f ? v : 0.f;
        C[(size_t)(m_base + m * 16 + sel * 8 + r) * N + col] = v;
      }
    }
  }
}

// ---------------------------------------------------------------------------
// f3 (1024 -> 10) + log_softmax, one wave per batch row.
// ---------------------------------------------------------------------------
__global__ __launch_bounds__(32)
void f3_logsoftmax(const float* __restrict__ X, const float* __restrict__ W,
                   const float* __restrict__ B, float* __restrict__ out)
{
  const int row = blockIdx.x;
  const int lane = threadIdx.x;
  float acc[10] = {};
  for (int k = lane; k < 1024; k += 32) {
    float x = X[(size_t)row * 1024 + k];
    #pragma unroll
    for (int c = 0; c < 10; ++c) acc[c] += x * W[k * 10 + c];
  }
  #pragma unroll
  for (int c = 0; c < 10; ++c) {
    #pragma unroll
    for (int off = 16; off > 0; off >>= 1)
      acc[c] += __shfl_xor(acc[c], off, 32);
  }
  if (lane == 0) {
    float z[10], m = -1e30f;
    #pragma unroll
    for (int c = 0; c < 10; ++c) { z[c] = acc[c] + B[c]; m = fmaxf(m, z[c]); }
    float s = 0.f;
    #pragma unroll
    for (int c = 0; c < 10; ++c) s += __expf(z[c] - m);
    float ls = __logf(s);
    #pragma unroll
    for (int c = 0; c < 10; ++c) out[row * 10 + c] = z[c] - m - ls;
  }
}

// ---------------------------------------------------------------------------
// Workspace layout (bytes)
// ---------------------------------------------------------------------------
#define OFF_SWG1 ((size_t)0)
#define OFF_SWG2 (OFF_SWG1 + 32 * 1024)            // g1: 1x32 tiles * 1KB
#define OFF_SWG3 (OFF_SWG2 + 512 * 1024)           // g2: 16x32 tiles
#define OFF_SWG4 (OFF_SWG3 + 512 * 1024)
#define OFF_SWF1 (OFF_SWG4 + 512 * 1024)
#define OFF_SWF2 (OFF_SWF1 + 1024 * 1024)          // f1: 16x64 tiles
#define OFF_XG   (OFF_SWF2 + 2048 * 1024)          // f2: 32x64 tiles
#define OFF_XF1  (OFF_XG + (size_t)1024 * 512 * 4)
#define OFF_XF2  (OFF_XF1 + (size_t)1024 * 1024 * 4)

extern "C" void kernel_launch(void* const* d_in, const int* in_sizes, int n_in,
                              void* d_out, int out_size, void* d_ws, size_t ws_size,
                              hipStream_t stream)
{
  const float* img = (const float*)d_in[0];
  const float* qst = (const float*)d_in[1];
  const float* g1w = (const float*)d_in[2];  const float* g1b = (const float*)d_in[3];
  const float* g2w = (const float*)d_in[4];  const float* g2b = (const float*)d_in[5];
  const float* g3w = (const float*)d_in[6];  const float* g3b = (const float*)d_in[7];
  const float* g4w = (const float*)d_in[8];  const float* g4b = (const float*)d_in[9];
  const float* f1w = (const float*)d_in[10]; const float* f1b = (const float*)d_in[11];
  const float* f2w = (const float*)d_in[12]; const float* f2b = (const float*)d_in[13];
  const float* f3w = (const float*)d_in[14]; const float* f3b = (const float*)d_in[15];

  char* ws = (char*)d_ws;
  __bf16* swg1 = (__bf16*)(ws + OFF_SWG1);
  __bf16* swg2 = (__bf16*)(ws + OFF_SWG2);
  __bf16* swg3 = (__bf16*)(ws + OFF_SWG3);
  __bf16* swg4 = (__bf16*)(ws + OFF_SWG4);
  __bf16* swf1 = (__bf16*)(ws + OFF_SWF1);
  __bf16* swf2 = (__bf16*)(ws + OFF_SWF2);
  float*  xg   = (float*)(ws + OFF_XG);
  float*  xf1  = (float*)(ws + OFF_XF1);
  float*  xf2  = (float*)(ws + OFF_XF2);

  // Per-launch weight swizzle (deterministic, re-done every call)
  swizzle_w<<<64,   256, 0, stream>>>(g1w, swg1, 32,   512, 23);
  swizzle_w<<<1024, 256, 0, stream>>>(g2w, swg2, 512,  512, 512);
  swizzle_w<<<1024, 256, 0, stream>>>(g3w, swg3, 512,  512, 512);
  swizzle_w<<<1024, 256, 0, stream>>>(g4w, swg4, 512,  512, 512);
  swizzle_w<<<2048, 256, 0, stream>>>(f1w, swf1, 512,  1024, 512);
  swizzle_w<<<4096, 256, 0, stream>>>(f2w, swf2, 1024, 1024, 1024);
  zero_f32<<<(1024 * 512 + 255) / 256, 256, 0, stream>>>(xg, 1024 * 512);

  // Fused g-MLP + sum-pool: 264 KB dynamic LDS per workgroup
  size_t g_lds = (size_t)(128 * 32 + 2 * 128 * 512) * 2;
  g_mlp_pool<<<dim3(5, 1024), 256, g_lds, stream>>>(
      img, qst, swg1, g1b, swg2, g2b, swg3, g3b, swg4, g4b, xg);

  // f-MLP
  gemm_f<<<8, 256, (size_t)128 * 512 * 2,  stream>>>(xg,  swf1, f1b, xf1, 512,  1024, 1);
  gemm_f<<<8, 256, (size_t)128 * 1024 * 2, stream>>>(xf1, swf2, f2b, xf2, 1024, 1024, 1);
  f3_logsoftmax<<<1024, 32, 0, stream>>>(xf2, f3w, f3b, (float*)d_out);

  (void)in_sizes; (void)n_in; (void)out_size; (void)ws_size;
}
